// RNN_82824149336294
// MI455X (gfx1250) — matrix operand order repaired and checked by reading, not compiled, not run on hardware
//
#include <hip/hip_runtime.h>
#include <hip/hip_bf16.h>
#include <math.h>

// Problem constants (match reference): SEQ, BATCH, IN, HID
constexpr int S   = 2048;
constexpr int B   = 128;
constexpr int DIN = 64;
constexpr int H   = 256;

typedef __attribute__((ext_vector_type(16))) __bf16 v16bf;
typedef __attribute__((ext_vector_type(8)))  __bf16 v8bf;
typedef __attribute__((ext_vector_type(8)))  float  v8f;
typedef __attribute__((ext_vector_type(4)))  unsigned int u32x4;
typedef __attribute__((ext_vector_type(8)))  int i32x8;
typedef __attribute__((ext_vector_type(4)))  int i32x4;

#define SHUF16(lo, hi) __builtin_shufflevector((lo), (hi), 0,1,2,3,4,5,6,7,8,9,10,11,12,13,14,15)

// tanh(x) = 1 - 2/(exp(2x)+1): native v_exp_f32 (TRANS, co-executes with XDL
// WMMA) + rcp-based fast divide. Exact saturation at +/-1.
__device__ __forceinline__ float fast_tanh(float x) {
  float e = __expf(2.0f * x);
  return 1.0f - __fdividef(2.0f, e + 1.0f);
}

// ---------------------------------------------------------------------------
// Kernel W: convert Wh (f32, row-major [H][H]) to bf16 in d_ws for the TDM.
// ---------------------------------------------------------------------------
__global__ __launch_bounds__(256) void wh_to_bf16(const float* __restrict__ Wh,
                                                  __bf16* __restrict__ whb) {
  const int i = blockIdx.x * 256 + threadIdx.x;
  whb[i] = (__bf16)Wh[i];
}

// ---------------------------------------------------------------------------
// Kernel A: xp[s*b][n] = sum_k x[s*b][k]*Wx[n][k] + bx[n] + bh[n]
// (bh folded into the bias: the scan's per-step bias add disappears.)
// One block per 16-row M-tile of the flattened (S*B, DIN) input. Memory-bound.
// Writes directly into d_out (h_seq region); the scan updates it in place.
// ---------------------------------------------------------------------------
__global__ __launch_bounds__(256) void rnn_xproj(const float* __restrict__ x,
                                                 const float* __restrict__ Wx,
                                                 const float* __restrict__ bx,
                                                 const float* __restrict__ bh,
                                                 float* __restrict__ xp) {
  __shared__ __bf16 sWx[H * DIN];   // 32 KB
  __shared__ __bf16 sX[16 * DIN];   //  2 KB

  const int tid  = threadIdx.x;
  const int lane = tid & 31;
  const int wave = tid >> 5;       // 8 waves; wave handles ntiles {2w, 2w+1}
  const int lm   = lane & 15;

  for (int i = tid; i < H * DIN; i += 256) sWx[i] = (__bf16)Wx[i];
  const long rowBase = (long)blockIdx.x * 16;
  for (int i = tid; i < 16 * DIN; i += 256) sX[i] = (__bf16)x[rowBase * DIN + i];
  __syncthreads();

  const int kaoff = (lane < 16) ? 0 : 16;   // A-frag K sub-base per half-wave
  const int kboff = (lane < 16) ? 0 : 16;   // B-frag K sub-base per half-wave

  v8f acc[2];
  #pragma unroll
  for (int j = 0; j < 2; ++j) {
    const int n = (wave * 2 + j) * 16 + lm;
    const float bias = bx[n] + bh[n];
    #pragma unroll
    for (int r = 0; r < 8; ++r) acc[j][r] = bias;
  }

  #pragma unroll
  for (int kt = 0; kt < 2; ++kt) {          // K = 64 -> two 32-deep WMMA steps
    const int kb = kt * 32 + kaoff;
    v8bf lo = *(const v8bf*)&sX[lm * DIN + kb];
    v8bf hi = *(const v8bf*)&sX[lm * DIN + kb + 8];
    v16bf a = SHUF16(lo, hi);
    #pragma unroll
    for (int j = 0; j < 2; ++j) {
      const int n = (wave * 2 + j) * 16 + lm;
      v16bf bf = *(const v16bf*)&sWx[n * DIN + kt * 32 + kboff];
      acc[j] = __builtin_amdgcn_wmma_f32_16x16x32_bf16(false, a, false, bf,
                                                       (short)0, acc[j], false, false);
    }
  }

  #pragma unroll
  for (int j = 0; j < 2; ++j) {
    const int n = (wave * 2 + j) * 16 + lm;
    #pragma unroll
    for (int r = 0; r < 8; ++r) {
      const int m = (lane < 16) ? r : (r + 8);
      xp[(rowBase + m) * H + n] = acc[j][r];
    }
  }
}

// ---------------------------------------------------------------------------
// Kernel B: sequential scan  h = tanh(xp_t + h @ Wh^T)   (biases pre-folded)
// Single 1024-thread workgroup (32 waves = WGP max). Wh bf16 in LDS (128 KB,
// DMA'd in by the Tensor Data Mover when d_ws is available) + double-buffered
// bf16 h (2 x 64 KB) = 256 KB of CDNA5's 320 KB WGP LDS. Each wave owns 4 of
// the 128 output 16x16 tiles -> 32 v_wmma_f32_16x16x32_bf16 per wave per step.
// xp[t+1] is preloaded into registers during step t's WMMA chain so each step
// starts on LDS-resident data only; global_prefetch warms L2 for t+2.
// ---------------------------------------------------------------------------
__global__ __launch_bounds__(1024) void rnn_scan(const float* __restrict__ Wh,
                                                 const __bf16* __restrict__ whb, // bf16 Wh in d_ws, or null
                                                 float* __restrict__ seq,        // in: xp, out: h_seq
                                                 float* __restrict__ hlast) {
  extern __shared__ char smem[];
  __bf16* sWh = (__bf16*)smem;                       // [H][H]  128 KB @ LDS offset 0
  __bf16* h0  = (__bf16*)(smem + (size_t)H * H * 2); // [B][H]   64 KB
  __bf16* h1  = h0 + B * H;                          // [B][H]   64 KB

  const int tid   = threadIdx.x;
  const int lane  = tid & 31;
  const int wave  = tid >> 5;
  const int mtile = wave & 7;          // 8 M-tiles cover B=128 rows
  const int nbase = (wave >> 3) * 4;   // 4 consecutive N-tiles per wave
  const int lm    = lane & 15;
  const int rsel  = (lane < 16) ? 0 : 8;
  const int kaoff = (lane < 16) ? 0 : 16;
  const int kboff = (lane < 16) ? 0 : 16;

  if (whb != nullptr) {
    // TDM bulk DMA: one tensor_load_to_lds moves the whole 256x256 bf16 Wh
    // tile from global into LDS, overlapped with the h-buffer zeroing below.
    if (tid == 0) {
      const unsigned long long ga = (unsigned long long)(uintptr_t)whb;
      u32x4 g0;
      g0[0] = 1u;                                        // count=1 user descriptor
      g0[1] = 0u;                                        // lds_addr = 0 (sWh at base)
      g0[2] = (unsigned)(ga & 0xFFFFFFFFu);              // global_addr[31:0]
      g0[3] = (unsigned)((ga >> 32) & 0x01FFFFFFu)       // global_addr[56:32]
            | 0x80000000u;                               // type=2 ("image")
      i32x8 g1;
      g1[0] = (int)(1u << 16);       // workgroup_mask=0, data_size=1 (2 bytes)
      g1[1] = (int)((unsigned)H << 16);   // tensor_dim0[15:0] @ bits[63:48]
      g1[2] = (int)((unsigned)H << 16);   // tensor_dim1[15:0] @ bits[95:80]
      g1[3] = (int)((unsigned)H << 16);   // tile_dim0 @ bits[127:112]
      g1[4] = (int)H;                     // tile_dim1 @ bits[143:128]; tile_dim2=0
      g1[5] = (int)H;                     // tensor_dim0_stride[31:0]
      g1[6] = 0;                          // stride hi / tensor_dim1_stride
      g1[7] = 0;
      i32x4 gz4 = {0, 0, 0, 0};           // groups 2/3 unused (2-D tensor)
      i32x8 gz8 = {0, 0, 0, 0, 0, 0, 0, 0};
      __builtin_amdgcn_tensor_load_to_lds(g0, g1, gz4, gz4, gz8, 0);
    }
  } else {
    // Fallback (d_ws too small): convert in-kernel.
    for (int i = tid; i < H * H; i += 1024) sWh[i] = (__bf16)Wh[i];
  }

  for (int i = tid; i < B * H; i += 1024) { h0[i] = (__bf16)0.0f; h1[i] = (__bf16)0.0f; }

  if (whb != nullptr && tid == 0) __builtin_amdgcn_s_wait_tensorcnt(0);
  __syncthreads();

  __bf16* hcur = h0;
  __bf16* hnxt = h1;

  // Preload xp[0] into registers.
  v8f xreg[4];
  #pragma unroll
  for (int j = 0; j < 4; ++j) {
    const int n = (nbase + j) * 16 + lm;
    #pragma unroll
    for (int r = 0; r < 8; ++r) {
      const int m = mtile * 16 + r + rsel;
      xreg[j][r] = seq[m * H + n];
    }
  }

  for (int t = 0; t < S; ++t) {
    float* xt = seq + (long)t * (B * H);

    // C = preloaded xp_t (bias already folded in by rnn_xproj)
    v8f acc[4];
    #pragma unroll
    for (int j = 0; j < 4; ++j) acc[j] = xreg[j];

    // h @ Wh^T : K = 256 -> 8 WMMA K-steps, A fragment reused across 4 N-tiles
    #pragma unroll
    for (int kt = 0; kt < 8; ++kt) {
      const int kb    = kt * 32;
      const int abase = (mtile * 16 + lm) * H + kb + kaoff;
      v8bf lo = *(const v8bf*)&hcur[abase];
      v8bf hi = *(const v8bf*)&hcur[abase + 8];
      v16bf a = SHUF16(lo, hi);
      #pragma unroll
      for (int j = 0; j < 4; ++j) {
        const int n = (nbase + j) * 16 + lm;
        v16bf bf = *(const v16bf*)&sWh[n * H + kb + kboff];
        acc[j] = __builtin_amdgcn_wmma_f32_16x16x32_bf16(false, a, false, bf,
                                                         (short)0, acc[j], false, false);
      }
    }

    // Preload xp[t+1] during this step's tail; its writers (step t+1, same
    // wave, same tiles) only run after these reads complete -> race-free.
    if (t + 1 < S) {
      const float* xt1 = xt + B * H;
      #pragma unroll
      for (int j = 0; j < 4; ++j) {
        const int n = (nbase + j) * 16 + lm;
        #pragma unroll
        for (int r = 0; r < 8; ++r) {
          const int m = mtile * 16 + r + rsel;
          xreg[j][r] = xt1[m * H + n];
        }
      }
    }
    // Warm L2 two steps ahead (1024 threads x 128B covers the 128 KB slice).
    if (t + 2 < S)
      __builtin_prefetch(seq + (long)(t + 2) * (B * H) + (size_t)tid * 32, 0, 1);

    // tanh, write h_seq (f32, in place over xp) + next-step bf16 h in LDS
    #pragma unroll
    for (int j = 0; j < 4; ++j) {
      const int n = (nbase + j) * 16 + lm;
      #pragma unroll
      for (int r = 0; r < 8; ++r) {
        const int m = mtile * 16 + r + rsel;
        const float v = fast_tanh(acc[j][r]);
        xt[m * H + n]   = v;
        hnxt[m * H + n] = (__bf16)v;
        if (t == S - 1) hlast[m * H + n] = v;
      }
    }

    __syncthreads();
    __bf16* tmp = hcur; hcur = hnxt; hnxt = tmp;
  }
}

// ---------------------------------------------------------------------------
extern "C" void kernel_launch(void* const* d_in, const int* in_sizes, int n_in,
                              void* d_out, int out_size, void* d_ws, size_t ws_size,
                              hipStream_t stream) {
  const float* x  = (const float*)d_in[0];
  const float* Wx = (const float*)d_in[1];
  const float* bx = (const float*)d_in[2];
  const float* Wh = (const float*)d_in[3];
  const float* bh = (const float*)d_in[4];

  float* out   = (float*)d_out;
  float* hseq  = out;                          // (S, B, H)
  float* hlast = out + (size_t)S * B * H;      // (B, H)

  const size_t lds = (size_t)(H * H + 2 * B * H) * sizeof(__bf16); // 256 KB
  (void)hipFuncSetAttribute((const void*)rnn_scan,
                            hipFuncAttributeMaxDynamicSharedMemorySize, (int)lds);

  // TDM path requires ws >= 128 KB for the bf16 Wh staging buffer.
  const bool tdm_ok = ws_size >= (size_t)H * H * sizeof(__bf16);
  __bf16* whb = tdm_ok ? (__bf16*)d_ws : nullptr;

  if (tdm_ok)
    wh_to_bf16<<<dim3(H * H / 256), dim3(256), 0, stream>>>(Wh, whb);

  rnn_xproj<<<dim3((S * B) / 16), dim3(256), 0, stream>>>(x, Wx, bx, bh, hseq);

  rnn_scan<<<dim3(1), dim3(1024), lds, stream>>>(Wh, whb, hseq, hlast);

  (void)in_sizes; (void)n_in; (void)out_size;
}
